// AttentionBlock_76647986364612
// MI455X (gfx1250) — matrix operand order repaired and checked by reading, not compiled, not run on hardware
//
#include <hip/hip_runtime.h>

#define DEVINL __device__ __forceinline__

typedef _Float16 half_t;
typedef __attribute__((ext_vector_type(16))) _Float16 v16h;
typedef __attribute__((ext_vector_type(8)))  _Float16 v8h;
typedef __attribute__((ext_vector_type(8)))  float    v8f;

constexpr int kC  = 512;
constexpr int kB  = 2;
constexpr int kHW = 4096;
constexpr int kG  = 32;
constexpr float kEPS = 1e-6f;

DEVINL int lane_id() { return (int)(threadIdx.x & 31u); }

// ---- WMMA fragment helpers (gfx1250 wave32 16x16x32 f16 layouts) ----------
// A (16x32, MxK), row-major A with leading dim lda (halves):
//   lane l: row m = l&15 ; kh = l>>4 ; holds K = {k0+kh*16+e, k0+kh*16+8+e}
DEVINL v16h load_a_frag(const half_t* __restrict__ A, int lda, int m0, int k0) {
  const int l  = lane_id();
  const int m  = m0 + (l & 15);
  const int kh = (l >> 4) & 1;
  const half_t* p = A + (size_t)m * lda + (size_t)(k0 + kh * 16);
  v8h lo = *(const v8h*)(p);
  v8h hi = *(const v8h*)(p + 8);
  v16h r;
#pragma unroll
  for (int e = 0; e < 8; ++e) { r[e] = lo[e]; r[8 + e] = hi[e]; }
  return r;
}

DEVINL v8f wmma_f16(v16h a, v16h b, v8f c) {
  return __builtin_amdgcn_wmma_f32_16x16x32_f16(false, a, false, b, (short)0, c,
                                                false, false);
}

// ---- CDNA5 async copy global -> LDS (ASYNCcnt-tracked) --------------------
DEVINL uint32_t lds_off_of(const void* p) {
  // flat shared-aperture address: low 32 bits are the LDS byte address
  return (uint32_t)(uintptr_t)p;
}

DEVINL void async_copy_b128(uint32_t lds_off, const half_t* gsrc) {
  asm volatile("global_load_async_to_lds_b128 %0, %1, off"
               :: "v"(lds_off), "v"((uint64_t)(uintptr_t)gsrc)
               : "memory");
}

DEVINL void wait_async_le2() { asm volatile("s_wait_asynccnt 0x2" ::: "memory"); }
DEVINL void wait_async_le0() { asm volatile("s_wait_asynccnt 0x0" ::: "memory"); }

// ---- shared GEMM mainloop --------------------------------------------------
// Block tile: 256(M) x 64(N); 8 waves, each wave owns rows [m0, m0+32)
// (two 16-row A subtiles => 8 independent WMMA chains per k-step).
// B supplied as B^T row-major Bt[n][k] (ldb halves). B tile staged to LDS by
// all 256 threads with async copies, double-buffered, KCHUNK=64.
struct AccT { v8f a[2][4]; };

template <int KTOT>
DEVINL AccT gemm_ldsB(const half_t* __restrict__ A, int lda, int m0,
                      const half_t* __restrict__ Bt, int ldb, int n0blk,
                      half_t* sB /* [2][64*64] halves */) {
  const int t   = (int)threadIdx.x;   // 0..255
  const int row = t >> 2;             // 0..63 : B^T row (column n)
  const int ch  = t & 3;              // this thread stages chunks ch and ch+4
  const half_t* gsrcRow = Bt + (size_t)(n0blk + row) * ldb;

  constexpr int NST = KTOT / 64;

  // stage(buf, k0): copy Bt[n0blk..+63][k0..k0+63] into sB[buf]
  auto stage = [&](int buf, int k0) {
    half_t* dstRow = sB + buf * (64 * 64) + row * 64;
    async_copy_b128(lds_off_of(dstRow + ch * 8),       gsrcRow + k0 + ch * 8);
    async_copy_b128(lds_off_of(dstRow + (ch + 4) * 8), gsrcRow + k0 + (ch + 4) * 8);
  };

  AccT ac;
#pragma unroll
  for (int u = 0; u < 2; ++u)
#pragma unroll
    for (int v = 0; v < 4; ++v) ac.a[u][v] = v8f{};

  const int l   = lane_id();
  const int bro = (l & 15);           // column within 16-wide subtile
  const int kko = (l >> 4) * 16;      // lane-half K offset

  stage(0, 0);
  for (int s = 0; s < NST; ++s) {
    if (s + 1 < NST) { stage((s + 1) & 1, (s + 1) * 64); wait_async_le2(); }
    else             { wait_async_le0(); }
    __syncthreads();          // B tile for stage s visible to all waves

    const half_t* bb = sB + (s & 1) * (64 * 64);
#pragma unroll
    for (int ks = 0; ks < 64; ks += 32) {
      const int k = s * 64 + ks;
      v16h a0 = load_a_frag(A, lda, m0,      k);
      v16h a1 = load_a_frag(A, lda, m0 + 16, k);
      v16h b0 = *(const v16h*)(bb + (size_t)( 0 + bro) * 64 + ks + kko);
      v16h b1 = *(const v16h*)(bb + (size_t)(16 + bro) * 64 + ks + kko);
      v16h b2 = *(const v16h*)(bb + (size_t)(32 + bro) * 64 + ks + kko);
      v16h b3 = *(const v16h*)(bb + (size_t)(48 + bro) * 64 + ks + kko);
      ac.a[0][0] = wmma_f16(a0, b0, ac.a[0][0]);
      ac.a[0][1] = wmma_f16(a0, b1, ac.a[0][1]);
      ac.a[0][2] = wmma_f16(a0, b2, ac.a[0][2]);
      ac.a[0][3] = wmma_f16(a0, b3, ac.a[0][3]);
      ac.a[1][0] = wmma_f16(a1, b0, ac.a[1][0]);
      ac.a[1][1] = wmma_f16(a1, b1, ac.a[1][1]);
      ac.a[1][2] = wmma_f16(a1, b2, ac.a[1][2]);
      ac.a[1][3] = wmma_f16(a1, b3, ac.a[1][3]);
    }
    __syncthreads();          // all waves done reading before overwrite
  }
  return ac;
}

// ---- elementwise f32 -> f16 -----------------------------------------------
__global__ void cvt_f32_to_f16(const float* __restrict__ src,
                               half_t* __restrict__ dst, int n) {
  int i = blockIdx.x * blockDim.x + threadIdx.x;
  if (i < n) dst[i] = (half_t)src[i];
}

// ---- GroupNorm, writes token-major f16 hnT[b][i][c] -----------------------
__global__ void groupnorm_t(const float* __restrict__ x,
                            const float* __restrict__ gamma,
                            const float* __restrict__ beta,
                            half_t* __restrict__ hnT) {
  const int bg = blockIdx.x;          // b*G + g
  const int b  = bg / kG, g = bg % kG;
  const int CG = kC / kG;             // 16 channels per group
  const int N  = CG * kHW;            // 65536 elems per group
  const float* xg = x + ((size_t)b * kC + (size_t)g * CG) * kHW;

  float s = 0.f, s2 = 0.f;
  for (int i = threadIdx.x; i < N; i += blockDim.x) {
    float v = xg[i]; s += v; s2 += v * v;
  }
  __shared__ float r1[256], r2[256];
  r1[threadIdx.x] = s; r2[threadIdx.x] = s2;
  __syncthreads();
  for (int off = 128; off > 0; off >>= 1) {
    if ((int)threadIdx.x < off) {
      r1[threadIdx.x] += r1[threadIdx.x + off];
      r2[threadIdx.x] += r2[threadIdx.x + off];
    }
    __syncthreads();
  }
  const float mu   = r1[0] / (float)N;
  const float var  = r2[0] / (float)N - mu * mu;
  const float rstd = rsqrtf(var + kEPS);

  for (int i = threadIdx.x; i < N; i += blockDim.x) {
    const int cl = i / kHW, t = i % kHW;
    const int c  = g * CG + cl;
    const float v = (xg[i] - mu) * rstd * gamma[c] + beta[c];
    hnT[((size_t)b * kHW + t) * kC + c] = (half_t)v;
  }
}

// ---- QKV GEMM: D(m=channel, n=token); M=512, N=4096, K=512 ----------------
__global__ void qkv_gemm(const half_t* __restrict__ W,
                         const float* __restrict__ bias,
                         const half_t* __restrict__ hnT,
                         half_t* __restrict__ outT,   // token-major [b][i][c]
                         half_t* __restrict__ outC,   // channel-major [b][c][i]
                         float scale) {
  __shared__ __align__(32) half_t sB[2 * 64 * 64];
  const int MB = kC / 256;                    // 2
  const int NB = kHW / 64;                    // 64
  const int blk = blockIdx.x;
  const int b    = blk / (MB * NB);
  const int rem  = blk % (MB * NB);
  const int mblk = rem / NB;
  const int nblk = rem % NB;
  const int w  = (int)(threadIdx.x >> 5);
  const int m0 = mblk * 256 + w * 32;
  const int n0 = nblk * 64;
  const half_t* Bt = hnT + (size_t)b * kHW * kC;

  AccT ac = gemm_ldsB<kC>(W, kC, m0, Bt, kC, n0, sB);

  const int l = lane_id();
#pragma unroll
  for (int u = 0; u < 2; ++u) {
    const int mb = m0 + u * 16 + 8 * (l >> 4);
#pragma unroll
    for (int t = 0; t < 4; ++t) {
      v8f acc = ac.a[u][t];
      const int n = n0 + t * 16 + (l & 15);
      if (outT) {
        half_t* p = outT + ((size_t)b * kHW + n) * kC + mb;
#pragma unroll
        for (int q = 0; q < 8; ++q)
          p[q] = (half_t)((acc[q] + bias[mb + q]) * scale);
      } else {
        half_t* p = outC + ((size_t)b * kC + mb) * kHW + n;
#pragma unroll
        for (int q = 0; q < 8; ++q)
          p[(size_t)q * kHW] = (half_t)((acc[q] + bias[mb + q]) * scale);
      }
    }
  }
}

// ---- S = (q*scale)^T k : M=N=4096, K=512, f32 out -------------------------
__global__ void s_gemm(const half_t* __restrict__ qT,
                       const half_t* __restrict__ kT,
                       float* __restrict__ S) {
  __shared__ __align__(32) half_t sB[2 * 64 * 64];
  const int MB = kHW / 256;                   // 16
  const int NB = kHW / 64;                    // 64
  const int blk = blockIdx.x;
  const int b    = blk / (MB * NB);
  const int rem  = blk % (MB * NB);
  const int mblk = rem / NB;
  const int nblk = rem % NB;
  const int w  = (int)(threadIdx.x >> 5);
  const int m0 = mblk * 256 + w * 32;
  const int n0 = nblk * 64;
  const half_t* Aq = qT + (size_t)b * kHW * kC;
  const half_t* Bk = kT + (size_t)b * kHW * kC;

  AccT ac = gemm_ldsB<kC>(Aq, kC, m0, Bk, kC, n0, sB);

  const int l = lane_id();
  float* Sb = S + (size_t)b * kHW * kHW;
#pragma unroll
  for (int u = 0; u < 2; ++u) {
    const int mb = m0 + u * 16 + 8 * (l >> 4);
#pragma unroll
    for (int t = 0; t < 4; ++t) {
      v8f acc = ac.a[u][t];
      const int n = n0 + t * 16 + (l & 15);
      float* p = Sb + (size_t)mb * kHW + n;
#pragma unroll
      for (int q = 0; q < 8; ++q) p[(size_t)q * kHW] = acc[q];
    }
  }
}

// ---- row softmax: one block per row, writes f16 P -------------------------
__global__ void softmax_rows(const float* __restrict__ S,
                             half_t* __restrict__ P) {
  const size_t row = blockIdx.x;              // b*HW + i
  const float* s = S + row * kHW;
  half_t* p = P + row * kHW;
  __shared__ float red[256];

  float m = -3.4e38f;
  for (int j = threadIdx.x; j < kHW; j += 256) m = fmaxf(m, s[j]);
  red[threadIdx.x] = m; __syncthreads();
  for (int off = 128; off > 0; off >>= 1) {
    if ((int)threadIdx.x < off)
      red[threadIdx.x] = fmaxf(red[threadIdx.x], red[threadIdx.x + off]);
    __syncthreads();
  }
  m = red[0];
  __syncthreads();

  float sum = 0.f;
  for (int j = threadIdx.x; j < kHW; j += 256) sum += __expf(s[j] - m);
  red[threadIdx.x] = sum; __syncthreads();
  for (int off = 128; off > 0; off >>= 1) {
    if ((int)threadIdx.x < off) red[threadIdx.x] += red[threadIdx.x + off];
    __syncthreads();
  }
  const float inv = 1.f / red[0];
  for (int j = threadIdx.x; j < kHW; j += 256)
    p[j] = (half_t)(__expf(s[j] - m) * inv);
}

// ---- O^T: hattn[c][i] = sum_j v[c][j] P[i][j]; store token-major [i][c] ---
// GEMM M=c (A=V row-major ld=HW), N=i (Bt=P row-major ld=HW), K=j=4096.
__global__ void o_gemm(const half_t* __restrict__ V,
                       const half_t* __restrict__ P,
                       half_t* __restrict__ OT) {
  __shared__ __align__(32) half_t sB[2 * 64 * 64];
  const int MB = kC / 256;                    // 2
  const int NB = kHW / 64;                    // 64
  const int blk = blockIdx.x;
  const int b    = blk / (MB * NB);
  const int rem  = blk % (MB * NB);
  const int mblk = rem / NB;
  const int nblk = rem % NB;
  const int w  = (int)(threadIdx.x >> 5);
  const int m0 = mblk * 256 + w * 32;
  const int n0 = nblk * 64;
  const half_t* Av = V + (size_t)b * kC * kHW;
  const half_t* Bp = P + (size_t)b * kHW * kHW;

  AccT ac = gemm_ldsB<kHW>(Av, kHW, m0, Bp, kHW, n0, sB);

  const int l = lane_id();
#pragma unroll
  for (int u = 0; u < 2; ++u) {
    const int mb = m0 + u * 16 + 8 * (l >> 4);
#pragma unroll
    for (int t = 0; t < 4; ++t) {
      v8f acc = ac.a[u][t];
      const int n = n0 + t * 16 + (l & 15);
      half_t* p = OT + ((size_t)b * kHW + n) * kC + mb;  // token-major [i][c]
#pragma unroll
      for (int q = 0; q < 8; ++q) p[q] = (half_t)acc[q];
    }
  }
}

// ---- out = wp * hattn + bp + x (f32) --------------------------------------
__global__ void out_gemm(const half_t* __restrict__ W,
                         const float* __restrict__ bias,
                         const half_t* __restrict__ hT,   // token-major [b][i][c]
                         const float* __restrict__ x,
                         float* __restrict__ out) {
  __shared__ __align__(32) half_t sB[2 * 64 * 64];
  const int MB = kC / 256;                    // 2
  const int NB = kHW / 64;                    // 64
  const int blk = blockIdx.x;
  const int b    = blk / (MB * NB);
  const int rem  = blk % (MB * NB);
  const int mblk = rem / NB;
  const int nblk = rem % NB;
  const int w  = (int)(threadIdx.x >> 5);
  const int m0 = mblk * 256 + w * 32;
  const int n0 = nblk * 64;
  const half_t* Bt = hT + (size_t)b * kHW * kC;

  AccT ac = gemm_ldsB<kC>(W, kC, m0, Bt, kC, n0, sB);

  const int l = lane_id();
#pragma unroll
  for (int u = 0; u < 2; ++u) {
    const int mb = m0 + u * 16 + 8 * (l >> 4);
#pragma unroll
    for (int t = 0; t < 4; ++t) {
      v8f acc = ac.a[u][t];
      const int n = n0 + t * 16 + (l & 15);
      float* po = out + ((size_t)b * kC + mb) * kHW + n;
      const float* px = x + ((size_t)b * kC + mb) * kHW + n;
#pragma unroll
      for (int q = 0; q < 8; ++q)
        po[(size_t)q * kHW] = acc[q] + bias[mb + q] + px[(size_t)q * kHW];
    }
  }
}

// ---------------------------------------------------------------------------
extern "C" void kernel_launch(void* const* d_in, const int* in_sizes, int n_in,
                              void* d_out, int out_size, void* d_ws, size_t ws_size,
                              hipStream_t stream) {
  (void)in_sizes; (void)n_in; (void)out_size; (void)ws_size;
  const float* x     = (const float*)d_in[0];
  const float* gamma = (const float*)d_in[1];
  const float* beta  = (const float*)d_in[2];
  const float* wq = (const float*)d_in[3];
  const float* bq = (const float*)d_in[4];
  const float* wk = (const float*)d_in[5];
  const float* bk = (const float*)d_in[6];
  const float* wv = (const float*)d_in[7];
  const float* bv = (const float*)d_in[8];
  const float* wp = (const float*)d_in[9];
  const float* bp = (const float*)d_in[10];
  float* out = (float*)d_out;

  char* ws = (char*)d_ws;
  size_t off = 0;
  auto alloc = [&](size_t bytes) -> void* {
    void* p = ws + off;
    off = (off + bytes + 255) & ~(size_t)255;
    return p;
  };
  half_t* wq_h = (half_t*)alloc((size_t)kC * kC * sizeof(half_t));
  half_t* wk_h = (half_t*)alloc((size_t)kC * kC * sizeof(half_t));
  half_t* wv_h = (half_t*)alloc((size_t)kC * kC * sizeof(half_t));
  half_t* wp_h = (half_t*)alloc((size_t)kC * kC * sizeof(half_t));
  half_t* hnT  = (half_t*)alloc((size_t)kB * kHW * kC * sizeof(half_t));
  half_t* qT   = (half_t*)alloc((size_t)kB * kHW * kC * sizeof(half_t));
  half_t* kT   = (half_t*)alloc((size_t)kB * kHW * kC * sizeof(half_t));
  half_t* vC   = (half_t*)alloc((size_t)kB * kC * kHW * sizeof(half_t));
  half_t* hT   = (half_t*)alloc((size_t)kB * kHW * kC * sizeof(half_t));
  float*  S    = (float*) alloc((size_t)kB * kHW * kHW * sizeof(float));
  half_t* P    = (half_t*)alloc((size_t)kB * kHW * kHW * sizeof(half_t));

  const int WC = kC * kC;
  cvt_f32_to_f16<<<(WC + 255) / 256, 256, 0, stream>>>(wq, wq_h, WC);
  cvt_f32_to_f16<<<(WC + 255) / 256, 256, 0, stream>>>(wk, wk_h, WC);
  cvt_f32_to_f16<<<(WC + 255) / 256, 256, 0, stream>>>(wv, wv_h, WC);
  cvt_f32_to_f16<<<(WC + 255) / 256, 256, 0, stream>>>(wp, wp_h, WC);

  groupnorm_t<<<kB * kG, 256, 0, stream>>>(x, gamma, beta, hnT);

  const float qscale = 0.044194173824159216f;  // 512^-0.5
  const int projBlocks = kB * (kC / 256) * (kHW / 64);     // 256
  qkv_gemm<<<projBlocks, 256, 0, stream>>>(wq_h, bq, hnT, qT, nullptr, qscale);
  qkv_gemm<<<projBlocks, 256, 0, stream>>>(wk_h, bk, hnT, kT, nullptr, 1.0f);
  qkv_gemm<<<projBlocks, 256, 0, stream>>>(wv_h, bv, hnT, nullptr, vC, 1.0f);

  const int sBlocks = kB * (kHW / 256) * (kHW / 64);       // 2048
  s_gemm<<<sBlocks, 256, 0, stream>>>(qT, kT, S);

  softmax_rows<<<kB * kHW, 256, 0, stream>>>(S, P);

  o_gemm<<<projBlocks, 256, 0, stream>>>(vC, P, hT);

  out_gemm<<<projBlocks, 256, 0, stream>>>(wp_h, bp, hT, x, out);
}
